// BiLSTM_CRF_49778670961380
// MI455X (gfx1250) — matrix operand order repaired and checked by reading, not compiled, not run on hardware
//
#include <hip/hip_runtime.h>
#include <hip/hip_bf16.h>

// ---------------------------------------------------------------------------
// BiLSTM-CRF NLL for gfx1250 (MI455X).  All GEMMs via v_wmma_f32_16x16x32_f16.
// Shapes: B=32, S=256, 2E=512, H=256 (per dir), 4H=1024, T=32.
// ---------------------------------------------------------------------------

typedef __attribute__((ext_vector_type(16))) _Float16 v16h;
typedef __attribute__((ext_vector_type(8)))  _Float16 v8h;
typedef __attribute__((ext_vector_type(8)))  float    v8f;

#define S_LEN 256
#define BATCH 32
#define EMB   512     // 2E
#define HDIM  256     // per-direction hidden
#define G4    1024    // 4H
#define NTAG  32

__device__ __forceinline__ float sigmoid_f(float x) {
    return 1.0f / (1.0f + __expf(-x));
}
__device__ __forceinline__ float tanh_f(float x) {
    float ax = fabsf(x);
    float t  = __expf(-2.0f * ax);
    float r  = (1.0f - t) / (1.0f + t);
    return x >= 0.0f ? r : -r;
}
__device__ __forceinline__ v16h pack16(v8h lo, v8h hi) {
    v16h r;
#pragma unroll
    for (int i = 0; i < 8; ++i) { r[i] = lo[i]; r[i + 8] = hi[i]; }
    return r;
}

// --------------------------- prep kernels ----------------------------------
__global__ __launch_bounds__(256) void k_f32_to_f16(const float* __restrict__ s,
                                                    _Float16* __restrict__ d, int n) {
    int i = blockIdx.x * 256 + threadIdx.x;
    if (i < n) d[i] = (_Float16)s[i];
}

__global__ __launch_bounds__(256) void k_bias_sum(const float* __restrict__ a,
                                                  const float* __restrict__ b,
                                                  float* __restrict__ o, int n) {
    int i = blockIdx.x * 256 + threadIdx.x;
    if (i < n) o[i] = a[i] + b[i];
}

// Embedding gather + concat -> f16 [B*S, 512]
__global__ __launch_bounds__(256) void k_embed(const int* __restrict__ tokens,
                                               const int* __restrict__ pos,
                                               const float* __restrict__ tok_emb,
                                               const float* __restrict__ pos_emb,
                                               _Float16* __restrict__ embh) {
    int bs = blockIdx.x;          // 0..8191  (= b*256 + s)
    int t  = threadIdx.x;         // 0..255
    int tk = tokens[bs];
    int pp = pos[bs];
    embh[bs * EMB + t]       = (_Float16)tok_emb[tk * 256 + t];
    embh[bs * EMB + 256 + t] = (_Float16)pos_emb[pp * 256 + t];
}

// --------------------------- input GEMM ------------------------------------
// C[m,n] = sum_k A[m,k]*W[n,k] + bias[n]   A:[8192,512]f16  W:[1024,512]f16
// One wave per 16x16 tile, K stepped by 32.
__global__ __launch_bounds__(256) void k_gemm_xg(const _Float16* __restrict__ A,
                                                 const _Float16* __restrict__ W,
                                                 const float* __restrict__ bias,
                                                 _Float16* __restrict__ C) {
    const int K = EMB, N = G4, NT = N / 16;                   // NT = 64
    int gw   = (blockIdx.x * blockDim.x + threadIdx.x) >> 5;  // global wave id
    int lane = threadIdx.x & 31;
    int tm = gw / NT, tn = gw % NT;
    int sel = lane >> 4, l15 = lane & 15;
    const _Float16* arow = A + (tm * 16 + l15) * K;
    const _Float16* wrow = W + (tn * 16 + l15) * K;
    v8f acc = {};
#pragma unroll 4
    for (int k0 = 0; k0 < K; k0 += 32) {
        // A fragment (16x32 f16 layout, ISA 7.12.2)
        v8h a0 = *(const v8h*)(arow + k0 + sel * 8);
        v8h a1 = *(const v8h*)(arow + k0 + 16 + sel * 8);
        // B fragment (32x16): lane = column, same K order as A
        v8h b0 = *(const v8h*)(wrow + k0 + sel * 8);
        v8h b1 = *(const v8h*)(wrow + k0 + 16 + sel * 8);
        acc = __builtin_amdgcn_wmma_f32_16x16x32_f16(false, pack16(a0, a1),
                                                     false, pack16(b0, b1),
                                                     (short)0, acc, false, false);
    }
#pragma unroll
    for (int r = 0; r < 8; ++r) {
        int row = tm * 16 + r + sel * 8;
        int col = tn * 16 + l15;
        C[row * N + col] = (_Float16)(acc[r] + bias[col]);
    }
}

// --------------------------- LSTM recurrence -------------------------------
// grid.x = 2 (direction), 1024 threads = 32 waves on one WGP.
// Wave (mtile, hgrp) keeps i/f/g/o accumulators and cell state c in registers
// for its 16 batches x 16 hidden columns; h is exchanged through LDS.
__global__ __launch_bounds__(1024) void k_lstm(const _Float16* __restrict__ Whh_f,
                                               const _Float16* __restrict__ Whh_r,
                                               const _Float16* __restrict__ Xg_f,
                                               const _Float16* __restrict__ Xg_r,
                                               _Float16* __restrict__ Hcat) {
    __shared__ _Float16 h_lds[BATCH][HDIM];   // 16 KB
    const int dir = blockIdx.x;
    const _Float16* W  = dir ? Whh_r : Whh_f;  // [1024,256] f16
    const _Float16* Xg = dir ? Xg_r  : Xg_f;   // [B*S,1024] f16 (b_ih+b_hh folded)

    int tid  = threadIdx.x;
    int lane = tid & 31;
    int wave = tid >> 5;
    int mtile = wave & 1;        // batch half
    int hgrp  = wave >> 1;       // 16-wide hidden-column group
    int sel = lane >> 4, l15 = lane & 15;
    int m0 = mtile * 16;
    int h0 = hgrp * 16;

    for (int i = tid; i < BATCH * HDIM; i += 1024)
        ((_Float16*)h_lds)[i] = (_Float16)0.0f;
    __syncthreads();

    v8f c = {};                               // cell state, register-resident
    const _Float16* hrow = &h_lds[m0 + l15][0];

    for (int t = 0; t < S_LEN; ++t) {
        int t_eff = dir ? (S_LEN - 1 - t) : t;   // time index into Xg
        v8f acc[4] = {};
        for (int k0 = 0; k0 < HDIM; k0 += 32) {
            v8h a0 = *(const v8h*)(hrow + k0 + sel * 8);
            v8h a1 = *(const v8h*)(hrow + k0 + 16 + sel * 8);
            v16h a = pack16(a0, a1);
#pragma unroll
            for (int g = 0; g < 4; ++g) {
                const _Float16* wrow = W + (g * HDIM + h0 + l15) * HDIM;
                __builtin_prefetch(wrow + k0 + 64, 0, 1);
                v8h b0 = *(const v8h*)(wrow + k0 + sel * 8);
                v8h b1 = *(const v8h*)(wrow + k0 + 16 + sel * 8);
                acc[g] = __builtin_amdgcn_wmma_f32_16x16x32_f16(false, a,
                             false, pack16(b0, b1), (short)0, acc[g], false, false);
            }
        }
        // gates + state update (in registers, C-fragment layout)
        _Float16 hnew[8];
#pragma unroll
        for (int r = 0; r < 8; ++r) {
            int b   = m0 + r + sel * 8;
            int col = h0 + l15;
            const _Float16* xg = Xg + (b * S_LEN + t_eff) * G4;
            float ig = sigmoid_f(acc[0][r] + (float)xg[0 * HDIM + col]);
            float fg = sigmoid_f(acc[1][r] + (float)xg[1 * HDIM + col]);
            float gg = tanh_f  (acc[2][r] + (float)xg[2 * HDIM + col]);
            float og = sigmoid_f(acc[3][r] + (float)xg[3 * HDIM + col]);
            float cn = fg * c[r] + ig * gg;
            c[r] = cn;
            hnew[r] = (_Float16)(og * tanh_f(cn));
        }
        __syncthreads();   // everyone done reading old h
        int s_out = dir ? (S_LEN - 1 - t) : t;
#pragma unroll
        for (int r = 0; r < 8; ++r) {
            int b = m0 + r + sel * 8;
            h_lds[b][h0 + l15] = hnew[r];
            Hcat[(b * S_LEN + s_out) * EMB + dir * HDIM + h0 + l15] = hnew[r];
        }
        __syncthreads();   // new h visible
    }
}

// --------------------------- emission GEMM ---------------------------------
// Em[m,n] = sum_k Hcat[m,k]*Wo[n,k] + bo[n]   M=8192, N=32, K=512
__global__ __launch_bounds__(256) void k_gemm_emis(const _Float16* __restrict__ A,
                                                   const _Float16* __restrict__ W,
                                                   const float* __restrict__ bias,
                                                   float* __restrict__ C) {
    const int K = EMB, N = NTAG, NT = N / 16;                 // NT = 2
    int gw   = (blockIdx.x * blockDim.x + threadIdx.x) >> 5;
    int lane = threadIdx.x & 31;
    int tm = gw / NT, tn = gw % NT;
    int sel = lane >> 4, l15 = lane & 15;
    const _Float16* arow = A + (tm * 16 + l15) * K;
    const _Float16* wrow = W + (tn * 16 + l15) * K;
    v8f acc = {};
#pragma unroll 4
    for (int k0 = 0; k0 < K; k0 += 32) {
        v8h a0 = *(const v8h*)(arow + k0 + sel * 8);
        v8h a1 = *(const v8h*)(arow + k0 + 16 + sel * 8);
        v8h b0 = *(const v8h*)(wrow + k0 + sel * 8);
        v8h b1 = *(const v8h*)(wrow + k0 + 16 + sel * 8);
        acc = __builtin_amdgcn_wmma_f32_16x16x32_f16(false, pack16(a0, a1),
                                                     false, pack16(b0, b1),
                                                     (short)0, acc, false, false);
    }
#pragma unroll
    for (int r = 0; r < 8; ++r) {
        int row = tm * 16 + r + sel * 8;
        int col = tn * 16 + l15;
        C[row * N + col] = acc[r] + bias[col];
    }
}

// --------------------------- CRF NLL ---------------------------------------
// One block, 32 waves <-> 32 batches, lane = tag (wave32-native).
__global__ __launch_bounds__(1024) void k_crf(const float* __restrict__ Em,   // [B,S,T]
                                              const int* __restrict__ labels, // [B,S]
                                              const float* __restrict__ start_tr,
                                              const float* __restrict__ end_tr,
                                              const float* __restrict__ trans, // [T,T]
                                              float* __restrict__ out) {
    __shared__ float tr[NTAG * NTAG];
    __shared__ float partial[BATCH];
    int tid = threadIdx.x;
    int b = tid >> 5, j = tid & 31;
    tr[tid] = trans[tid];          // 1024 threads load 32*32 entries
    __syncthreads();

    const float* em = Em + b * S_LEN * NTAG;
    float alpha = start_tr[j] + em[j];           // t = 0
    for (int t = 1; t < S_LEN; ++t) {
        float m = -1e30f;
#pragma unroll 8
        for (int i = 0; i < NTAG; ++i) {
            float ai = __shfl(alpha, i, 32);
            m = fmaxf(m, ai + tr[i * NTAG + j]);
        }
        float ssum = 0.0f;
#pragma unroll 8
        for (int i = 0; i < NTAG; ++i) {
            float ai = __shfl(alpha, i, 32);
            ssum += __expf(ai + tr[i * NTAG + j] - m);
        }
        alpha = m + __logf(ssum) + em[t * NTAG + j];
    }
    // log_z = logsumexp_j(alpha + end_tr[j])
    float v = alpha + end_tr[j];
    float m = v;
    for (int o = 16; o; o >>= 1) m = fmaxf(m, __shfl_xor(m, o, 32));
    float e = __expf(v - m);
    for (int o = 16; o; o >>= 1) e += __shfl_xor(e, o, 32);
    float log_z = m + __logf(e);

    // gold-path score
    const int* lab = labels + b * S_LEN;
    float sc = 0.0f;
    for (int s = j; s < S_LEN; s += 32)     sc += em[s * NTAG + lab[s]];
    for (int s = j; s < S_LEN - 1; s += 32) sc += tr[lab[s] * NTAG + lab[s + 1]];
    for (int o = 16; o; o >>= 1) sc += __shfl_xor(sc, o, 32);

    if (j == 0) {
        sc += start_tr[lab[0]] + end_tr[lab[S_LEN - 1]];
        partial[b] = log_z - sc;               // per-batch NLL contribution
    }
    __syncthreads();
    if (tid == 0) {
        float tot = 0.0f;
        for (int i = 0; i < BATCH; ++i) tot += partial[i];
        out[0] = tot;                          // -sum(score - log_z)
    }
}

// --------------------------- orchestration ---------------------------------
extern "C" void kernel_launch(void* const* d_in, const int* in_sizes, int n_in,
                              void* d_out, int out_size, void* d_ws, size_t ws_size,
                              hipStream_t stream) {
    (void)in_sizes; (void)n_in; (void)out_size; (void)ws_size;
    const int*   tokens   = (const int*)  d_in[0];
    const int*   pos_tags = (const int*)  d_in[1];
    const int*   labels   = (const int*)  d_in[2];
    const float* tok_emb  = (const float*)d_in[3];
    const float* pos_emb  = (const float*)d_in[4];
    const float* w_ih_f   = (const float*)d_in[5];
    const float* w_hh_f   = (const float*)d_in[6];
    const float* b_ih_f   = (const float*)d_in[7];
    const float* b_hh_f   = (const float*)d_in[8];
    const float* w_ih_b   = (const float*)d_in[9];
    const float* w_hh_b   = (const float*)d_in[10];
    const float* b_ih_b   = (const float*)d_in[11];
    const float* b_hh_b   = (const float*)d_in[12];
    const float* w_out    = (const float*)d_in[13];
    const float* b_out    = (const float*)d_in[14];
    const float* start_tr = (const float*)d_in[15];
    const float* end_tr   = (const float*)d_in[16];
    const float* trans    = (const float*)d_in[17];

    char* ws = (char*)d_ws;
    size_t off = 0;
    auto alloc = [&](size_t bytes) -> void* {
        void* p = ws + off;
        off += (bytes + 255) & ~(size_t)255;
        return p;
    };
    _Float16* embh   = (_Float16*)alloc((size_t)8192 * EMB * 2);   // 8 MB
    _Float16* wihf16 = (_Float16*)alloc((size_t)G4 * EMB * 2);     // 1 MB
    _Float16* wihb16 = (_Float16*)alloc((size_t)G4 * EMB * 2);
    _Float16* whhf16 = (_Float16*)alloc((size_t)G4 * HDIM * 2);    // 0.5 MB
    _Float16* whhb16 = (_Float16*)alloc((size_t)G4 * HDIM * 2);
    _Float16* wout16 = (_Float16*)alloc((size_t)NTAG * EMB * 2);
    float*    biasf  = (float*)   alloc((size_t)G4 * 4);
    float*    biasb  = (float*)   alloc((size_t)G4 * 4);
    _Float16* xgf    = (_Float16*)alloc((size_t)8192 * G4 * 2);    // 16 MB
    _Float16* xgb    = (_Float16*)alloc((size_t)8192 * G4 * 2);
    _Float16* hcat   = (_Float16*)alloc((size_t)8192 * EMB * 2);   // 8 MB
    float*    emis   = (float*)   alloc((size_t)8192 * NTAG * 4);  // 1 MB

    // weight conversion + bias folding
    k_f32_to_f16<<<(G4 * EMB + 255) / 256, 256, 0, stream>>>(w_ih_f, wihf16, G4 * EMB);
    k_f32_to_f16<<<(G4 * EMB + 255) / 256, 256, 0, stream>>>(w_ih_b, wihb16, G4 * EMB);
    k_f32_to_f16<<<(G4 * HDIM + 255) / 256, 256, 0, stream>>>(w_hh_f, whhf16, G4 * HDIM);
    k_f32_to_f16<<<(G4 * HDIM + 255) / 256, 256, 0, stream>>>(w_hh_b, whhb16, G4 * HDIM);
    k_f32_to_f16<<<(NTAG * EMB + 255) / 256, 256, 0, stream>>>(w_out, wout16, NTAG * EMB);
    k_bias_sum<<<(G4 + 255) / 256, 256, 0, stream>>>(b_ih_f, b_hh_f, biasf, G4);
    k_bias_sum<<<(G4 + 255) / 256, 256, 0, stream>>>(b_ih_b, b_hh_b, biasb, G4);

    // embedding gather
    k_embed<<<8192, 256, 0, stream>>>(tokens, pos_tags, tok_emb, pos_emb, embh);

    // input-projection GEMMs: 512 x 64 tiles = 32768 waves -> 4096 blocks
    k_gemm_xg<<<4096, 256, 0, stream>>>(embh, wihf16, biasf, xgf);
    k_gemm_xg<<<4096, 256, 0, stream>>>(embh, wihb16, biasb, xgb);

    // sequential BiLSTM scan: one workgroup per direction
    k_lstm<<<2, 1024, 0, stream>>>(whhf16, whhb16, xgf, xgb, hcat);

    // emissions: 512 x 2 tiles = 1024 waves -> 128 blocks
    k_gemm_emis<<<128, 256, 0, stream>>>(hcat, wout16, b_out, emis);

    // CRF forward algorithm + gold score + final scalar
    k_crf<<<1, 1024, 0, stream>>>(emis, labels, start_tr, end_tr, trans, (float*)d_out);
}
